// STGCNFineTunedModel_28716151341469
// MI455X (gfx1250) — matrix operand order repaired and checked by reading, not compiled, not run on hardware
//
#include <hip/hip_runtime.h>
#include <hip/hip_bf16.h>

// ---------------------------------------------------------------------------
// ST-GCN inference for MI455X (gfx1250, wave32, WMMA).
// All convolutions are lowered to bf16 WMMA GEMMs (f32 accumulate):
//   activations:  [m=(n,t,v)][c]  bf16, channel-minor (contiguous K)
//   weights:      packed bf16 [tap][o][c]
//   epilogue:     acc*scale[o] + shift[o] (+residual) (+relu) -> bf16
// Tiles are staged with CDNA5 async global->LDS copies (ASYNCcnt) in a
// double-buffered pipeline: stage s+1 copies overlap stage s WMMAs.
// ---------------------------------------------------------------------------

typedef __attribute__((ext_vector_type(16))) __bf16 v16bf;
typedef __attribute__((ext_vector_type(8)))  float  v8f;

union Frag  { v16bf v; uint4 q[2]; };
union Pack8 { __bf16 e[8]; uint4 q; };

#define NBATCH 64
#define V17    17

// Async global->LDS 128-bit copy (per lane). LDS address = low 32 bits of the
// generic pointer.
__device__ __forceinline__ void async_b128(const __bf16* g, const __bf16* l) {
  unsigned loff = (unsigned)(unsigned long long)l;
  asm volatile("global_load_async_to_lds_b128 %0, %1, off"
               :: "v"(loff), "v"(g)
               : "memory");
}
// Wait until <=N async copies outstanding (async loads complete in order).
__device__ __forceinline__ void wait_async0() {
  asm volatile("s_wait_asynccnt 0x0" ::: "memory");
}
__device__ __forceinline__ void wait_async2() {
  asm volatile("s_wait_asynccnt 0x2" ::: "memory");
}

// ---------------------------------------------------------------------------
// Fused BN prep: scale = g * rsqrt(v+eps); shift = b - m*scale (+ bias*scale)
// ---------------------------------------------------------------------------
__global__ void k_fuse_bn(const float* __restrict__ g, const float* __restrict__ b,
                          const float* __restrict__ m, const float* __restrict__ v,
                          const float* __restrict__ bias,
                          float* __restrict__ scale, float* __restrict__ shift, int n) {
  int i = blockIdx.x * blockDim.x + threadIdx.x;
  if (i >= n) return;
  float s = g[i] * rsqrtf(v[i] + 1e-5f);
  scale[i] = s;
  float sh = b[i] - m[i] * s;
  if (bias) sh += bias[i] * s;
  shift[i] = sh;
}

// ---------------------------------------------------------------------------
// Input: x[n][c][t][v] fp32 -> data_bn (per v*3+c channel) -> X0[(n,t,v)][32] bf16
// channels 3..31 zero padded.
// ---------------------------------------------------------------------------
__global__ __launch_bounds__(256) void k_input(const float* __restrict__ x,
                                               const float* __restrict__ sc,
                                               const float* __restrict__ sh,
                                               __bf16* __restrict__ X0) {
  int i = blockIdx.x * 256 + threadIdx.x;
  if (i >= NBATCH * 300 * V17 * 32) return;
  int c = i & 31;
  int rest = i >> 5;
  int v = rest % V17;
  int nt = rest / V17;
  int t = nt % 300, n = nt / 300;
  float val = 0.f;
  if (c < 3) {
    int ch = v * 3 + c;
    val = x[((n * 3 + c) * 300 + t) * V17 + v] * sc[ch] + sh[ch];
  }
  X0[i] = (__bf16)val;
}

// ---------------------------------------------------------------------------
// Graph mix: Y[(n,t,w)][c] = sum_v A[v][w] * X[(n,t,v)][c]   (17-wide, ~1% FLOPs)
// ---------------------------------------------------------------------------
__global__ __launch_bounds__(256) void k_amix(const __bf16* __restrict__ X,
                                              const float* __restrict__ A,
                                              __bf16* __restrict__ Y, int Cp, int NT) {
  __shared__ float As[289];
  for (int i = threadIdx.x; i < 289; i += 256) As[i] = A[i];
  __syncthreads();
  int idx = blockIdx.x * 256 + threadIdx.x;
  int total = NT * V17 * Cp;
  if (idx >= total) return;
  int c = idx % Cp;
  int w = (idx / Cp) % V17;
  int nt = idx / (Cp * V17);
  const __bf16* xr = X + nt * V17 * Cp + c;
  float acc = 0.f;
#pragma unroll
  for (int v = 0; v < V17; ++v) acc += As[v * V17 + w] * (float)xr[v * Cp];
  Y[idx] = (__bf16)acc;
}

// ---------------------------------------------------------------------------
// Weight packing to bf16.
// ---------------------------------------------------------------------------
__global__ void k_pack2d(const float* __restrict__ w, __bf16* __restrict__ out,
                         int Co, int Ci, int Cp) {  // w[Co][Ci] -> out[Co][Cp]
  int i = blockIdx.x * 256 + threadIdx.x;
  if (i >= Co * Cp) return;
  int c = i % Cp, o = i / Cp;
  out[i] = (c < Ci) ? (__bf16)w[o * Ci + c] : (__bf16)0.f;
}

__global__ void k_pack_tcn(const float* __restrict__ w, __bf16* __restrict__ out,
                           int Co) {  // w[o][c][9] -> out[k][o][c]
  int i = blockIdx.x * 256 + threadIdx.x;
  if (i >= 9 * Co * Co) return;
  int c = i % Co;
  int o = (i / Co) % Co;
  int k = i / (Co * Co);
  out[i] = (__bf16)w[(o * Co + c) * 9 + k];
}

// ---------------------------------------------------------------------------
// WMMA GEMM: Y[m_out][o] = sum_{k,c} W[k][o][c] * Xin[(n, t_out*stride+k-taps/2, v)][c]
// Block tile 64x64, BK=32; 8 waves, each wave two 16x16 v_wmma_f32_16x16x32_bf16
// tiles. Double-buffered async global->LDS staging; out-of-range taps are
// address-clamped on copy and zeroed per-lane in the B fragment.
// ---------------------------------------------------------------------------
__global__ __launch_bounds__(256) void k_gemm(
    const __bf16* __restrict__ W, const __bf16* __restrict__ Xin,
    __bf16* __restrict__ Yout,
    const float* __restrict__ scale, const float* __restrict__ shift,
    const __bf16* __restrict__ Res,
    int Co, int Cin, int T_in, int T_out, int stride, int taps, int relu) {
  __shared__ __bf16 As[2 * 64 * 32];  // ping-pong [o_local][k]
  __shared__ __bf16 Bs[2 * 64 * 32];  // ping-pong [col_local][k]

  const int tid = threadIdx.x;
  const int lane = tid & 31, wv = tid >> 5;
  const int tr = wv >> 1, tcc = wv & 1;      // wave -> (row tile, col pair)
  const int half = lane >> 4, lr = lane & 15;
  const int arow = tid >> 2, akc = (tid & 3) * 8;  // cooperative copy mapping
  const int obase = blockIdx.y * 64;
  const int koff = taps >> 1;

  // copy-thread B column -> (n, t_out, v)
  const int m_b = blockIdx.x * 64 + arow;
  const int vB = m_b % V17;
  const int tt = m_b / V17;
  const int tpB = tt % T_out;
  const int nB = tt / T_out;

  // per-lane validity sources for the two output columns of this wave
  const int mA = blockIdx.x * 64 + tcc * 32 + lr;        // column of tile 0
  const int tpL0 = (mA / V17) % T_out;
  const int tpL1 = ((mA + 16) / V17) % T_out;

  const int nc = Cin >> 5;                // K-chunks per tap (1,2,4,8)
  const int ncs = __builtin_ctz(nc);
  const int S = taps * nc;                // total pipeline stages

  v8f acc0 = {0.f, 0.f, 0.f, 0.f, 0.f, 0.f, 0.f, 0.f};
  v8f acc1 = {0.f, 0.f, 0.f, 0.f, 0.f, 0.f, 0.f, 0.f};

  // Issue the two async copies (A chunk + B chunk) for a stage.
  auto issue = [&](int st) {
    const int k = st >> ncs;
    const int c0 = (st & (nc - 1)) << 5;
    int t_in = tpB * stride + k - koff;
    t_in = t_in < 0 ? 0 : (t_in >= T_in ? T_in - 1 : t_in);  // clamp (zeroed later)
    const __bf16* ga = W + k * Co * Cin + (obase + arow) * Cin + c0 + akc;
    const __bf16* gb = Xin + ((nB * T_in + t_in) * V17 + vB) * Cin + c0 + akc;
    const int buf = (st & 1) * (64 * 32);
    async_b128(ga, &As[buf + arow * 32 + akc]);
    async_b128(gb, &Bs[buf + arow * 32 + akc]);
    if (c0 + 32 < Cin) __builtin_prefetch(gb + 32, 0, 1);
  };

  issue(0);
  for (int st = 0; st < S; ++st) {
    if (st + 1 < S) { issue(st + 1); wait_async2(); }  // stage st done, st+1 in flight
    else            { wait_async0(); }
    __syncthreads();

    const int buf = (st & 1) * (64 * 32);
    const int k = st >> ncs;

    // A frag: lane half 0 -> K {0..7,16..23}, half 1 -> K {8..15,24..31}
    Frag af, b0, b1;
    const int ar = tr * 16 + lr;
    af.q[0] = *(const uint4*)&As[buf + ar * 32 + half * 8];
    af.q[1] = *(const uint4*)&As[buf + ar * 32 + 16 + half * 8];
    // B frag: lane half 0 -> K {0..7,16..23} of its column, half 1 -> K {8..15,24..31}
    const int c0l = tcc * 32 + lr;
    b0.q[0] = *(const uint4*)&Bs[buf + c0l * 32 + half * 8];
    b0.q[1] = *(const uint4*)&Bs[buf + c0l * 32 + 16 + half * 8];
    b1.q[0] = *(const uint4*)&Bs[buf + (c0l + 16) * 32 + half * 8];
    b1.q[1] = *(const uint4*)&Bs[buf + (c0l + 16) * 32 + 16 + half * 8];

    // zero fragments of temporally out-of-range columns (per lane)
    if (taps > 1) {
      const uint4 z = {0u, 0u, 0u, 0u};
      const int t0 = tpL0 * stride + k - koff;
      const int t1 = tpL1 * stride + k - koff;
      if ((unsigned)t0 >= (unsigned)T_in) { b0.q[0] = z; b0.q[1] = z; }
      if ((unsigned)t1 >= (unsigned)T_in) { b1.q[0] = z; b1.q[1] = z; }
    }

    acc0 = __builtin_amdgcn_wmma_f32_16x16x32_bf16(false, af.v, false, b0.v,
                                                   (short)0, acc0, false, false);
    acc1 = __builtin_amdgcn_wmma_f32_16x16x32_bf16(false, af.v, false, b1.v,
                                                   (short)0, acc1, false, false);
    __syncthreads();  // all reads of buf done -> safe to overwrite at issue(st+2)
  }

  // Epilogue. D layout: VGPR r, lanes 0-15 -> M=r, lanes 16-31 -> M=r+8; N=lane%16.
  const int oloc = obase + tr * 16 + half * 8;
  float sc[8], sh[8];
#pragma unroll
  for (int r = 0; r < 8; ++r) { sc[r] = scale[oloc + r]; sh[r] = shift[oloc + r]; }
#pragma unroll
  for (int bt = 0; bt < 2; ++bt) {
    const v8f a = bt ? acc1 : acc0;
    const int mcol = blockIdx.x * 64 + tcc * 32 + bt * 16 + lr;
    Pack8 rv;
    if (Res) rv.q = *(const uint4*)(Res + mcol * Co + oloc);  // 128-bit residual load
    Pack8 pk;
#pragma unroll
    for (int r = 0; r < 8; ++r) {
      float vv = a[r] * sc[r] + sh[r];
      if (Res) vv += (float)rv.e[r];
      if (relu) vv = fmaxf(vv, 0.f);
      pk.e[r] = (__bf16)vv;
    }
    *(uint4*)&Yout[mcol * Co + oloc] = pk.q;
  }
}

// ---------------------------------------------------------------------------
// Global mean over (t,v) and final FC.
// ---------------------------------------------------------------------------
__global__ void k_mean(const __bf16* __restrict__ X, float* __restrict__ feat,
                       int TV, int Co) {
  int n = blockIdx.x, o = threadIdx.x;
  const __bf16* p = X + (size_t)n * TV * Co + o;
  float acc = 0.f;
  for (int i = 0; i < TV; ++i) acc += (float)p[(size_t)i * Co];
  feat[n * Co + o] = acc / (float)TV;
}

__global__ void k_fc(const float* __restrict__ feat, const float* __restrict__ fw,
                     const float* __restrict__ fb, float* __restrict__ out) {
  int i = blockIdx.x * blockDim.x + threadIdx.x;
  if (i >= 128) return;
  int n = i >> 1, j = i & 1;
  float acc = fb[j];
  for (int c = 0; c < 256; ++c) acc += feat[n * 256 + c] * fw[j * 256 + c];
  out[i] = acc;
}

// ---------------------------------------------------------------------------
// Host orchestration.
// ---------------------------------------------------------------------------
struct Cfg { int ci, co, stride, kind; };  // kind: 0=none, 1=id, 2=conv
static const Cfg g_cfgs[9] = {
  {3, 64, 1, 0},  {64, 64, 1, 1},  {64, 64, 1, 1},
  {64, 128, 2, 2},{128, 128, 1, 1},{128, 128, 1, 1},
  {128, 256, 2, 2},{256, 256, 1, 1},{256, 256, 1, 1}};

extern "C" void kernel_launch(void* const* d_in, const int* in_sizes, int n_in,
                              void* d_out, int out_size, void* d_ws, size_t ws_size,
                              hipStream_t stream) {
  (void)in_sizes; (void)n_in; (void)out_size; (void)ws_size;
  const float* const* fin = reinterpret_cast<const float* const*>(d_in);

  // jax tree-flatten order (sorted dict keys): A, blocks[0..8]{gcn_b, gcn_bn{b,g,m,v},
  // gcn_w, [res_b, res_bn{b,g,m,v}, res_w], tcn_b, tcn_bn{b,g,m,v}, tcn_w},
  // data_bn{b,g,m,v}, fc_b, fc_w, x.
  int ii = 0;
  const float* Aptr = fin[ii++];
  const float *g_b[9], *g_bn[9][4], *g_w[9];
  const float *r_b[9], *r_bn[9][4], *r_w[9];
  const float *t_b[9], *t_bn[9][4], *t_w[9];
  for (int b = 0; b < 9; ++b) {
    g_b[b] = fin[ii++];
    for (int j = 0; j < 4; ++j) g_bn[b][j] = fin[ii++];
    g_w[b] = fin[ii++];
    if (g_cfgs[b].kind == 2) {
      r_b[b] = fin[ii++];
      for (int j = 0; j < 4; ++j) r_bn[b][j] = fin[ii++];
      r_w[b] = fin[ii++];
    } else { r_b[b] = nullptr; r_w[b] = nullptr; }
    t_b[b] = fin[ii++];
    for (int j = 0; j < 4; ++j) t_bn[b][j] = fin[ii++];
    t_w[b] = fin[ii++];
  }
  const float* dbn[4];
  for (int j = 0; j < 4; ++j) dbn[j] = fin[ii++];
  const float* fcb = fin[ii++];
  const float* fcw = fin[ii++];
  const float* xin = fin[ii++];

  // Workspace layout (~210 MB).
  char* base = (char*)d_ws;
  size_t off = 0;
  auto give = [&](size_t bytes) -> void* {
    void* p = base + off;
    off = (off + bytes + 255) & ~(size_t)255;
    return p;
  };
  const size_t MAXACT = 20889600ULL;            // 64*300*17*64 elements
  __bf16* X0 = (__bf16*)give(MAXACT * 2);
  __bf16* X1 = (__bf16*)give(MAXACT * 2);
  __bf16* G  = (__bf16*)give(2 * MAXACT * 2);   // pre-stride GCN output
  __bf16* AR = (__bf16*)give(MAXACT * 2);       // A-mix buffer, reused as residual
  __bf16* Wg = (__bf16*)give(256 * 256 * 2);
  __bf16* Wr = (__bf16*)give(256 * 256 * 2);
  __bf16* Wt = (__bf16*)give(9 * 256 * 256 * 2);
  float* sG = (float*)give(256 * 4); float* hG = (float*)give(256 * 4);
  float* sT = (float*)give(256 * 4); float* hT = (float*)give(256 * 4);
  float* sR = (float*)give(256 * 4); float* hR = (float*)give(256 * 4);
  float* sD = (float*)give(64 * 4);  float* hD = (float*)give(64 * 4);
  float* feat = (float*)give(64 * 256 * 4);

  // data_bn + convert input (pad C 3->32)
  k_fuse_bn<<<1, 64, 0, stream>>>(dbn[1], dbn[0], dbn[2], dbn[3], nullptr, sD, hD, 51);
  k_input<<<(NBATCH * 300 * V17 * 32 + 255) / 256, 256, 0, stream>>>(xin, sD, hD, X0);

  __bf16* cur = X0;
  __bf16* nxt = X1;
  int T = 300;
  for (int b = 0; b < 9; ++b) {
    const int ci = g_cfgs[b].ci, co = g_cfgs[b].co;
    const int s = g_cfgs[b].stride, kind = g_cfgs[b].kind;
    const int Cin = (b == 0) ? 32 : ci;  // padded input channels
    const int Tout = T / s;
    const int Min = NBATCH * T * V17;    // divisible by 64 for T in {300,150,75}
    const int Mout = NBATCH * Tout * V17;

    k_pack2d<<<(co * Cin + 255) / 256, 256, 0, stream>>>(g_w[b], Wg, co, ci, Cin);
    k_fuse_bn<<<1, 256, 0, stream>>>(g_bn[b][1], g_bn[b][0], g_bn[b][2], g_bn[b][3],
                                     g_b[b], sG, hG, co);
    k_pack_tcn<<<(9 * co * co + 255) / 256, 256, 0, stream>>>(t_w[b], Wt, co);
    k_fuse_bn<<<1, 256, 0, stream>>>(t_bn[b][1], t_bn[b][0], t_bn[b][2], t_bn[b][3],
                                     t_b[b], sT, hT, co);
    if (kind == 2) {
      k_pack2d<<<(co * Cin + 255) / 256, 256, 0, stream>>>(r_w[b], Wr, co, ci, Cin);
      k_fuse_bn<<<1, 256, 0, stream>>>(r_bn[b][1], r_bn[b][0], r_bn[b][2], r_bn[b][3],
                                       r_b[b], sR, hR, co);
    }

    // 1) graph mix over V (on input channels)
    int amix_total = NBATCH * T * V17 * Cin;
    k_amix<<<(amix_total + 255) / 256, 256, 0, stream>>>(cur, Aptr, AR, Cin, NBATCH * T);

    // 2) GCN 1x1 GEMM + fused bias/BN -> G
    dim3 gg(Min / 64, co / 64);
    k_gemm<<<gg, 256, 0, stream>>>(Wg, AR, G, sG, hG, nullptr,
                                   co, Cin, T, T, 1, 1, 0);

    // 3) residual path for conv blocks (strided 1x1 GEMM + BN), reuses AR
    if (kind == 2) {
      dim3 gr(Mout / 64, co / 64);
      k_gemm<<<gr, 256, 0, stream>>>(Wr, cur, AR, sR, hR, nullptr,
                                     co, Cin, T, Tout, 2, 1, 0);
    }

    // 4) TCN 9-tap GEMM + fused bias/BN + residual + relu -> next activation
    const __bf16* resp = (kind == 1) ? cur : (kind == 2) ? AR : nullptr;
    dim3 gt(Mout / 64, co / 64);
    k_gemm<<<gt, 256, 0, stream>>>(Wt, G, nxt, sT, hT, resp,
                                   co, co, T, Tout, s, 9, 1);

    __bf16* tmp = cur; cur = nxt; nxt = tmp;
    T = Tout;
  }

  // mean over (t,v) then FC
  k_mean<<<64, 256, 0, stream>>>(cur, feat, T * V17, 256);
  k_fc<<<1, 128, 0, stream>>>(feat, fcw, fcb, (float*)d_out);
}
